// Prostate_58428735094818
// MI455X (gfx1250) — matrix operand order repaired and checked by reading, not compiled, not run on hardware
//
#include <hip/hip_runtime.h>
#include <hip/hip_bf16.h>
#include <math.h>

#define NN   50000
#define EE   300000
#define BB   32
#define DIMC 256
#define HID  1024   // 4*DIM
#define OUTC 4

typedef __attribute__((ext_vector_type(16))) __bf16       v16bf;
typedef __attribute__((ext_vector_type(8)))  float        v8f;
typedef __attribute__((ext_vector_type(8)))  unsigned int v8u;

union FragU { v8u u; v16bf b; };

// ---------------- helpers ----------------
__device__ inline unsigned short f2bf(float f) {
  unsigned int u = __float_as_uint(f);
  u += 0x7FFFu + ((u >> 16) & 1u);           // round-to-nearest-even
  return (unsigned short)(u >> 16);
}

__device__ inline void atomicMaxFloat(float* addr, float val) {
  if (val >= 0.0f) atomicMax((int*)addr, __float_as_int(val));
  else             atomicMin((unsigned int*)addr, __float_as_uint(val));
}

// A fragment (16x32 bf16) from a row-major 16x256 tile staged in LDS.
// lane<16: M=lane, K = k0+{0..15}
// lane>=16: same M, K = k0+{16..31}.
__device__ inline v16bf frag_a_lds(const unsigned short* A, int k0, int lane) {
  int m  = lane & 15;
  int kh = (lane >> 4) << 4;                 // 0 or 16
  const uint4* p = (const uint4*)(A + m * DIMC + k0 + kh);
  uint4 lo = p[0];                           // K .. K+7
  uint4 hi = p[1];                           // K+8 .. K+15
  FragU f; f.u = (v8u){lo.x, lo.y, lo.z, lo.w, hi.x, hi.y, hi.z, hi.w};
  return f.b;
}

// B fragment: 32x16 bf16 from pre-transposed weight Wt[n*DIMC + k].
// lane<16: col=lane, K=k0+{0..15};  lane>=16: col=lane-16, K=k0+{16..31}.
__device__ inline v16bf load_b_frag(const unsigned short* Wt, int c0, int k0, int lane) {
  int n  = lane & 15;
  int ks = k0 + ((lane >> 4) << 4);          // 0 or 16
  const uint4* p = (const uint4*)(Wt + (size_t)(c0 + n) * DIMC + ks);
  uint4 lo = p[0];
  uint4 hi = p[1];
  FragU f; f.u = (v8u){lo.x, lo.y, lo.z, lo.w, hi.x, hi.y, hi.z, hi.w};
  return f.b;
}

// ---------------- kernels ----------------
__global__ void fillf_k(float* p, float v, int n) {
  int i = blockIdx.x * blockDim.x + threadIdx.x;
  if (i < n) p[i] = v;
}

// transpose + bf16-convert one 256x256 weight: Wt[n*256+k] = bf16(W[k*256+n])
__global__ void wt_bf16_k(const float* __restrict__ W, unsigned short* __restrict__ Wt) {
  int tid = blockIdx.x * blockDim.x + threadIdx.x;     // 65536
  int k = tid >> 8, n = tid & 255;
  Wt[(size_t)n * DIMC + k] = f2bf(W[(size_t)k * DIMC + n]);
}

__global__ void degree_k(const int* __restrict__ dst, float* __restrict__ cnt) {
  int e = blockIdx.x * blockDim.x + threadIdx.x;
  if (e < EE) atomicAdd(&cnt[dst[e]], 1.0f);
}

__global__ void dinv_k(const float* __restrict__ cnt, float* __restrict__ dinv) {
  int n = blockIdx.x * blockDim.x + threadIdx.x;
  if (n < NN) dinv[n] = rsqrtf(cnt[n] + 1.0f);
}

// edge scatter-add: agg[dst] += xin[src], 4 floats per thread
__global__ void edge_scatter_k(const int* __restrict__ src, const int* __restrict__ dst,
                               const float* __restrict__ xin, float* __restrict__ agg) {
  int tid = blockIdx.x * blockDim.x + threadIdx.x;     // EE*64
  if (tid >= EE * 64) return;
  int e = tid >> 6;
  int c = (tid & 63) << 2;
  int s = src[e], d = dst[e];
  float4 v = *(const float4*)(xin + (size_t)s * DIMC + c);
  float* p = agg + (size_t)d * DIMC + c;
  atomicAdd(p + 0, v.x); atomicAdd(p + 1, v.y);
  atomicAdd(p + 2, v.z); atomicAdd(p + 3, v.w);
}

// Mb = bf16(agg / max(cnt,1)); Xb = bf16(xin)
__global__ void make_bf16_k(const float* __restrict__ agg, const float* __restrict__ cnt,
                            const float* __restrict__ xin,
                            unsigned short* __restrict__ Mb, unsigned short* __restrict__ Xb) {
  int tid = blockIdx.x * blockDim.x + threadIdx.x;     // NN*DIMC
  if (tid >= NN * DIMC) return;
  int node = tid >> 8;
  float c = fmaxf(cnt[node], 1.0f);
  Mb[tid] = f2bf(agg[tid] / c);
  Xb[tid] = f2bf(xin[tid]);
}

// h = mean@Wl + x@Wr + bl  via bf16 WMMA, f32 accumulate.
// One block = 128 threads (4 waves) owns a 16-row tile and ALL 16 column tiles:
// wave w computes column tiles {4w..4w+3}, reusing A fragments across them.
// The two 16x256 A tiles are staged into LDS with async global->LDS copies.
__global__ void sage_gemm_k(const unsigned short* __restrict__ Mb,
                            const unsigned short* __restrict__ Xb,
                            const unsigned short* __restrict__ Wlt,
                            const unsigned short* __restrict__ Wrt,
                            const float* __restrict__ bl,
                            float* __restrict__ h) {
  __shared__ unsigned short Asm[2][16 * DIMC];         // [0]=mean, [1]=x : 2 x 8KB

  int lane = threadIdx.x & 31;
  int wave = threadIdx.x >> 5;
  int r0 = blockIdx.x * 16;

  // ---- async stage: contiguous 8KB per matrix, 512 x 16B chunks, 4 per thread
  {
    const char* gm = (const char*)(Mb + (size_t)r0 * DIMC);
    const char* gx = (const char*)(Xb + (size_t)r0 * DIMC);
    unsigned int lm = (unsigned int)(size_t)(&Asm[0][0]);
    unsigned int lx = (unsigned int)(size_t)(&Asm[1][0]);
#pragma unroll
    for (int i = 0; i < 4; ++i) {
      int byteoff = (threadIdx.x + i * 128) * 16;
      asm volatile("global_load_async_to_lds_b128 %0, %1, off"
                   :: "v"(lm + byteoff), "v"(gm + byteoff) : "memory");
      asm volatile("global_load_async_to_lds_b128 %0, %1, off"
                   :: "v"(lx + byteoff), "v"(gx + byteoff) : "memory");
    }
    asm volatile("s_wait_asynccnt 0" ::: "memory");
  }
  __syncthreads();

  v8f acc0 = {}, acc1 = {}, acc2 = {}, acc3 = {};
  int cb = wave * 64;                                  // 64-column strip per wave
#pragma unroll
  for (int k0 = 0; k0 < DIMC; k0 += 32) {
    v16bf am = frag_a_lds(&Asm[0][0], k0, lane);
    v16bf ax = frag_a_lds(&Asm[1][0], k0, lane);

    v16bf b0 = load_b_frag(Wlt, cb +  0, k0, lane);
    acc0 = __builtin_amdgcn_wmma_f32_16x16x32_bf16(false, am, false, b0, (short)0, acc0, false, false);
    v16bf b1 = load_b_frag(Wrt, cb +  0, k0, lane);
    acc0 = __builtin_amdgcn_wmma_f32_16x16x32_bf16(false, ax, false, b1, (short)0, acc0, false, false);

    v16bf b2 = load_b_frag(Wlt, cb + 16, k0, lane);
    acc1 = __builtin_amdgcn_wmma_f32_16x16x32_bf16(false, am, false, b2, (short)0, acc1, false, false);
    v16bf b3 = load_b_frag(Wrt, cb + 16, k0, lane);
    acc1 = __builtin_amdgcn_wmma_f32_16x16x32_bf16(false, ax, false, b3, (short)0, acc1, false, false);

    v16bf b4 = load_b_frag(Wlt, cb + 32, k0, lane);
    acc2 = __builtin_amdgcn_wmma_f32_16x16x32_bf16(false, am, false, b4, (short)0, acc2, false, false);
    v16bf b5 = load_b_frag(Wrt, cb + 32, k0, lane);
    acc2 = __builtin_amdgcn_wmma_f32_16x16x32_bf16(false, ax, false, b5, (short)0, acc2, false, false);

    v16bf b6 = load_b_frag(Wlt, cb + 48, k0, lane);
    acc3 = __builtin_amdgcn_wmma_f32_16x16x32_bf16(false, am, false, b6, (short)0, acc3, false, false);
    v16bf b7 = load_b_frag(Wrt, cb + 48, k0, lane);
    acc3 = __builtin_amdgcn_wmma_f32_16x16x32_bf16(false, ax, false, b7, (short)0, acc3, false, false);
  }

  int rbase = r0 + ((lane >> 4) << 3);
  int nsub  = lane & 15;
  v8f accs[4] = {acc0, acc1, acc2, acc3};
#pragma unroll
  for (int t = 0; t < 4; ++t) {
    int col = cb + t * 16 + nsub;
    float bias = bl[col];
#pragma unroll
    for (int r = 0; r < 8; ++r)
      h[(size_t)(rbase + r) * DIMC + col] = accs[t][r] + bias;
  }
}

// global max pool into z0[b*HID + off + col]
__global__ void gmax_k(const float* __restrict__ h, const int* __restrict__ batch,
                       float* __restrict__ z0, int off) {
  int tid = blockIdx.x * blockDim.x + threadIdx.x;     // NN*DIMC
  if (tid >= NN * DIMC) return;
  int node = tid >> 8, col = tid & 255;
  atomicMaxFloat(&z0[(size_t)batch[node] * HID + off + col], h[tid]);
}

__global__ void gcn_xw_k(const float* __restrict__ h, const float* __restrict__ Wg,
                         float* __restrict__ xw) {
  int n = blockIdx.x * blockDim.x + threadIdx.x;
  if (n >= NN) return;
  float s = 0.0f;
  const float* row = h + (size_t)n * DIMC;
  for (int k = 0; k < DIMC; ++k) s += row[k] * Wg[k];
  xw[n] = s;
}

__global__ void sc_init_k(const float* __restrict__ dinv, const float* __restrict__ xw,
                          const float* __restrict__ bg, float* __restrict__ sraw) {
  int n = blockIdx.x * blockDim.x + threadIdx.x;
  if (n >= NN) return;
  sraw[n] = dinv[n] * dinv[n] * xw[n] + bg[0];
}

__global__ void sc_edge_k(const int* __restrict__ src, const int* __restrict__ dst,
                          const float* __restrict__ dinv, const float* __restrict__ xw,
                          float* __restrict__ sraw) {
  int e = blockIdx.x * blockDim.x + threadIdx.x;
  if (e >= EE) return;
  int s = src[e], d = dst[e];
  atomicAdd(&sraw[d], dinv[s] * dinv[d] * xw[s]);
}

__global__ void segmax_k(const float* __restrict__ sraw, const int* __restrict__ batch,
                         float* __restrict__ segmax) {
  int n = blockIdx.x * blockDim.x + threadIdx.x;
  if (n >= NN) return;
  atomicMaxFloat(&segmax[batch[n]], sraw[n]);
}

__global__ void seg_exp_k(const float* __restrict__ sraw, const int* __restrict__ batch,
                          const float* __restrict__ segmax, float* __restrict__ score,
                          float* __restrict__ segsum) {
  int n = blockIdx.x * blockDim.x + threadIdx.x;
  if (n >= NN) return;
  int b = batch[n];
  float e = __expf(sraw[n] - segmax[b]);
  score[n] = e;
  atomicAdd(&segsum[b], e);
}

// SAGPool masked max of (h * score) into z0[...,768+col]
__global__ void p3_k(const float* __restrict__ h, const int* __restrict__ batch,
                     const float* __restrict__ score, const float* __restrict__ segsum,
                     float* __restrict__ z0) {
  int tid = blockIdx.x * blockDim.x + threadIdx.x;     // NN*DIMC
  if (tid >= NN * DIMC) return;
  int node = tid >> 8, col = tid & 255;
  int b = batch[node];
  float ssum = segsum[b];
  float sc = score[node] / ssum;
  float thr = fminf(1.0f / ssum - 1e-7f, 0.001f);      // min(maxscore - TOL, MIN_SCORE)
  if (sc > thr)
    atomicMaxFloat(&z0[(size_t)b * HID + 768 + col], h[tid] * sc);
}

__global__ void fc1_k(const float* __restrict__ z0, const float* __restrict__ W1,
                      const float* __restrict__ b1, float* __restrict__ z1) {
  int tid = blockIdx.x * blockDim.x + threadIdx.x;     // BB*HID
  if (tid >= BB * HID) return;
  int b = tid >> 10, j = tid & 1023;
  float s = b1[j];
  const float* zr = z0 + (size_t)b * HID;
  for (int k = 0; k < HID; ++k) s += zr[k] * W1[(size_t)k * HID + j];
  z1[tid] = s;
}

__global__ void bn_relu_k(const float* __restrict__ z1, const float* __restrict__ gamma,
                          const float* __restrict__ beta, float* __restrict__ z2) {
  int j = blockIdx.x * blockDim.x + threadIdx.x;       // HID
  if (j >= HID) return;
  float mu = 0.0f;
  for (int b = 0; b < BB; ++b) mu += z1[(size_t)b * HID + j];
  mu *= (1.0f / BB);
  float var = 0.0f;
  for (int b = 0; b < BB; ++b) { float d = z1[(size_t)b * HID + j] - mu; var += d * d; }
  var *= (1.0f / BB);
  float inv = rsqrtf(var + 1e-5f);
  for (int b = 0; b < BB; ++b) {
    float v = (z1[(size_t)b * HID + j] - mu) * inv * gamma[j] + beta[j];
    z2[(size_t)b * HID + j] = fmaxf(v, 0.0f);
  }
}

__global__ void fc2_k(const float* __restrict__ z2, const float* __restrict__ W2,
                      const float* __restrict__ b2, float* __restrict__ out) {
  int tid = blockIdx.x * blockDim.x + threadIdx.x;     // BB*OUTC = 128
  if (tid >= BB * OUTC) return;
  int b = tid >> 2, o = tid & 3;
  float s = b2[o];
  const float* zr = z2 + (size_t)b * HID;
  for (int j = 0; j < HID; ++j) s += zr[j] * W2[(size_t)j * OUTC + o];
  out[tid] = s;
}

// one wave: per-row log-softmax + NLL, reduce, out[128] = loss
__global__ void loss_k(float* __restrict__ out, const int* __restrict__ y) {
  int b = threadIdx.x;                                 // 32 threads
  float l0 = out[b * 4 + 0], l1 = out[b * 4 + 1], l2 = out[b * 4 + 2], l3 = out[b * 4 + 3];
  float m = fmaxf(fmaxf(l0, l1), fmaxf(l2, l3));
  float lse = m + __logf(__expf(l0 - m) + __expf(l1 - m) + __expf(l2 - m) + __expf(l3 - m));
  float term = -(out[b * 4 + y[b]] - lse) * (1.0f / BB);
  for (int off = 16; off > 0; off >>= 1) term += __shfl_down(term, off, 32);
  if (b == 0) out[BB * OUTC] = term;
}

// ---------------- host ----------------
extern "C" void kernel_launch(void* const* d_in, const int* in_sizes, int n_in,
                              void* d_out, int out_size, void* d_ws, size_t ws_size,
                              hipStream_t stream) {
  const float* x     = (const float*)d_in[0];
  const int*   eidx  = (const int*)d_in[1];
  const int*   src   = eidx;
  const int*   dst   = eidx + EE;
  const int*   batch = (const int*)d_in[2];
  const int*   y     = (const int*)d_in[3];
  const float* Wl[3] = {(const float*)d_in[4], (const float*)d_in[7], (const float*)d_in[10]};
  const float* bl[3] = {(const float*)d_in[5], (const float*)d_in[8], (const float*)d_in[11]};
  const float* Wr[3] = {(const float*)d_in[6], (const float*)d_in[9], (const float*)d_in[12]};
  const float* Wg = (const float*)d_in[13];
  const float* bg = (const float*)d_in[14];
  const float* W1 = (const float*)d_in[15];
  const float* b1 = (const float*)d_in[16];
  const float* ga = (const float*)d_in[17];
  const float* be = (const float*)d_in[18];
  const float* W2 = (const float*)d_in[19];
  const float* b2 = (const float*)d_in[20];
  float* out = (float*)d_out;

  // workspace layout
  char* ws = (char*)d_ws;
  size_t off = 0;
  auto take = [&](size_t bytes) { char* p = ws + off; off += (bytes + 255) & ~(size_t)255; return p; };
  float*          cnt    = (float*)take((size_t)NN * 4);
  float*          dinv   = (float*)take((size_t)NN * 4);
  float*          agg    = (float*)take((size_t)NN * DIMC * 4);
  float*          hbuf   = (float*)take((size_t)NN * DIMC * 4);
  unsigned short* Xb     = (unsigned short*)take((size_t)NN * DIMC * 2);
  unsigned short* Mb     = (unsigned short*)take((size_t)NN * DIMC * 2);
  unsigned short* WT     = (unsigned short*)take((size_t)6 * DIMC * DIMC * 2);
  float*          xw     = (float*)take((size_t)NN * 4);
  float*          sraw   = (float*)take((size_t)NN * 4);
  float*          score  = (float*)take((size_t)NN * 4);
  float*          segmax = (float*)take((size_t)BB * 4);
  float*          segsum = (float*)take((size_t)BB * 4);
  float*          z0     = (float*)take((size_t)BB * HID * 4);
  float*          z1     = (float*)take((size_t)BB * HID * 4);
  float*          z2     = (float*)take((size_t)BB * HID * 4);
  (void)ws_size; (void)in_sizes; (void)n_in; (void)out_size;

  const int T = 256;
  // weight transpose + bf16 (Wl0,Wr0,Wl1,Wr1,Wl2,Wr2)
  for (int l = 0; l < 3; ++l) {
    wt_bf16_k<<<(DIMC * DIMC) / T, T, 0, stream>>>(Wl[l], WT + (size_t)(2 * l + 0) * DIMC * DIMC);
    wt_bf16_k<<<(DIMC * DIMC) / T, T, 0, stream>>>(Wr[l], WT + (size_t)(2 * l + 1) * DIMC * DIMC);
  }

  hipMemsetAsync(cnt, 0, (size_t)NN * 4, stream);
  degree_k<<<(EE + T - 1) / T, T, 0, stream>>>(dst, cnt);
  dinv_k<<<(NN + T - 1) / T, T, 0, stream>>>(cnt, dinv);

  fillf_k<<<(BB * HID + T - 1) / T, T, 0, stream>>>(z0, -INFINITY, BB * HID);

  const float* xin = x;
  for (int l = 0; l < 3; ++l) {
    hipMemsetAsync(agg, 0, (size_t)NN * DIMC * 4, stream);
    edge_scatter_k<<<(EE * 64 + T - 1) / T, T, 0, stream>>>(src, dst, xin, agg);
    make_bf16_k<<<(NN * DIMC + T - 1) / T, T, 0, stream>>>(agg, cnt, xin, Mb, Xb);
    sage_gemm_k<<<NN / 16, 128, 0, stream>>>(
        Mb, Xb, WT + (size_t)(2 * l + 0) * DIMC * DIMC, WT + (size_t)(2 * l + 1) * DIMC * DIMC,
        bl[l], hbuf);
    gmax_k<<<(NN * DIMC + T - 1) / T, T, 0, stream>>>(hbuf, batch, z0, l * DIMC);
    xin = hbuf;
  }

  // SAGPooling score path (GCNConv -> segment softmax)
  gcn_xw_k<<<(NN + T - 1) / T, T, 0, stream>>>(hbuf, Wg, xw);
  sc_init_k<<<(NN + T - 1) / T, T, 0, stream>>>(dinv, xw, bg, sraw);
  sc_edge_k<<<(EE + T - 1) / T, T, 0, stream>>>(src, dst, dinv, xw, sraw);
  fillf_k<<<1, BB, 0, stream>>>(segmax, -INFINITY, BB);
  fillf_k<<<1, BB, 0, stream>>>(segsum, 0.0f, BB);
  segmax_k<<<(NN + T - 1) / T, T, 0, stream>>>(sraw, batch, segmax);
  seg_exp_k<<<(NN + T - 1) / T, T, 0, stream>>>(sraw, batch, segmax, score, segsum);
  p3_k<<<(NN * DIMC + T - 1) / T, T, 0, stream>>>(hbuf, batch, score, segsum, z0);

  // MLP head
  fc1_k<<<(BB * HID + T - 1) / T, T, 0, stream>>>(z0, W1, b1, z1);
  bn_relu_k<<<(HID + T - 1) / T, T, 0, stream>>>(z1, ga, be, z2);
  fc2_k<<<1, BB * OUTC, 0, stream>>>(z2, W2, b2, out);
  loss_k<<<1, 32, 0, stream>>>(out, y);
}
